// ChainedGP_7885559955475
// MI455X (gfx1250) — matrix operand order repaired and checked by reading, not compiled, not run on hardware
//
#include <hip/hip_runtime.h>
#include <math.h>

#define N_TOT 16384
#define M_TOT 2048
#define VARF 1.0f
#define JITTERF 1e-6f
#define HALF_LOG_2PI_F 0.91893853320467274178f
#define NEG_HALF_INV_LS2 (-2.0f)   /* -0.5 / (0.5*0.5) */

typedef __attribute__((ext_vector_type(16))) _Float16 v16h;
typedef __attribute__((ext_vector_type(8)))  _Float16 v8h;
typedef __attribute__((ext_vector_type(8)))  float    v8f;

union H16 { v16h v; v8h h[2]; };

// ---- WMMA fragment loaders (all matrices have leading dim M_TOT) ----
// A 16x32 f16 layout: lane L: row = L&15; halves0..15 = K0..15 (+16 if L>=16).
__device__ inline v16h load_a_frag(const _Float16* A, int row0, int k0, int lane) {
    int r  = row0 + (lane & 15);
    int kh = (lane >> 4) * 16;
    const _Float16* p = A + (size_t)r * M_TOT + k0 + kh;
    H16 u;
    u.h[0] = *(const v8h*)(p);
    u.h[1] = *(const v8h*)(p + 8);
    return u.v;
}
// B 32x16 f16 from Bt = B^T row-major: lane L: col = L&15;
// halves0..15 = K = k0 + 16*(L>>4) + h  -> contiguous in Bt row.
__device__ inline v16h load_bt_frag(const _Float16* Bt, int col0, int k0, int lane) {
    int c  = col0 + (lane & 15);
    int kh = (lane >> 4) * 16;
    const _Float16* p = Bt + (size_t)c * M_TOT + k0 + kh;
    H16 u;
    u.h[0] = *(const v8h*)(p);
    u.h[1] = *(const v8h*)(p + 8);
    return u.v;
}
__device__ inline v8f wmma_f16(v16h a, v16h b, v8f c) {
    return __builtin_amdgcn_wmma_f32_16x16x32_f16(false, a, false, b, (short)0, c, false, false);
}
// store one 16x16 f32 accum tile as f16 at (r0, c0)
__device__ inline void store_tile(_Float16* D, int r0, int c0, int lane, v8f c) {
    int col = lane & 15, kh = lane >> 4;
#pragma unroll
    for (int v = 0; v < 8; ++v)
        D[(size_t)(r0 + v + 8 * kh) * M_TOT + c0 + col] = (_Float16)c[v];
}
// store tile transposed (for building W^T)
__device__ inline void store_tile_T(_Float16* D, int r0, int c0, int lane, v8f c) {
    int col = lane & 15, kh = lane >> 4;
#pragma unroll
    for (int v = 0; v < 8; ++v)
        D[(size_t)(c0 + col) * M_TOT + r0 + v + 8 * kh] = (_Float16)c[v];
}

// ---------------- utility ----------------
__global__ void k_fill0(float* p, int n) {
    int i = blockIdx.x * 256 + threadIdx.x;
    if (i < n) p[i] = 0.f;
}

// Kuu = rbf(z,z) + jitter*I
__global__ void k_kuu(const float* __restrict__ z, float* __restrict__ Kuu) {
    int j = blockIdx.x * 16 + threadIdx.x;
    int i = blockIdx.y * 16 + threadIdx.y;
    float d2 = 0.f;
#pragma unroll
    for (int d = 0; d < 8; ++d) { float t = z[i*8+d] - z[j*8+d]; d2 += t*t; }
    float v = VARF * expf(NEG_HALF_INV_LS2 * d2);
    if (i == j) v += JITTERF;
    Kuu[(size_t)i * M_TOT + j] = v;
}

// ---------------- blocked Cholesky (BS=64), f32 ----------------
__global__ void k_chol_diag(float* Kuu, int kb) {
    __shared__ float s[64][65];
    int tid = threadIdx.x, b = kb * 64;
    for (int idx = tid; idx < 4096; idx += 256)
        s[idx >> 6][idx & 63] = Kuu[(size_t)(b + (idx >> 6)) * M_TOT + b + (idx & 63)];
    __syncthreads();
    for (int j = 0; j < 64; ++j) {
        if (tid == 0) s[j][j] = sqrtf(s[j][j]);
        __syncthreads();
        float d = s[j][j];
        for (int i = j + 1 + tid; i < 64; i += 256) s[i][j] /= d;
        __syncthreads();
        for (int idx = tid; idx < 4096; idx += 256) {
            int r = idx >> 6, c = idx & 63;
            if (c > j && r >= c) s[r][c] -= s[r][j] * s[c][j];
        }
        __syncthreads();
    }
    for (int idx = tid; idx < 4096; idx += 256) {
        int r = idx >> 6, c = idx & 63;
        if (c <= r) Kuu[(size_t)(b + r) * M_TOT + b + c] = s[r][c];
    }
}

__global__ void k_chol_panel(float* Kuu, int kb) {
    __shared__ float Lkk[64][64];
    int tid = threadIdx.x, b = kb * 64, base = b + 64;
    for (int idx = tid; idx < 4096; idx += 256)
        Lkk[idx >> 6][idx & 63] = Kuu[(size_t)(b + (idx >> 6)) * M_TOT + b + (idx & 63)];
    __syncthreads();
    int r = base + blockIdx.x * 256 + tid;
    if (r >= M_TOT) return;
    float a[64];
#pragma unroll
    for (int j = 0; j < 64; ++j) a[j] = Kuu[(size_t)r * M_TOT + b + j];
#pragma unroll
    for (int j = 0; j < 64; ++j) {
        float v = a[j];
        for (int k2 = 0; k2 < j; ++k2) v -= a[k2] * Lkk[j][k2];
        a[j] = v / Lkk[j][j];
    }
#pragma unroll
    for (int j = 0; j < 64; ++j) Kuu[(size_t)r * M_TOT + b + j] = a[j];
}

__global__ void k_chol_syrk(float* Kuu, int kb) {
    if (blockIdx.x > blockIdx.y) return;   // lower triangle only
    __shared__ float pi[16][64], pj[16][64];
    int b = kb * 64, base = b + 64;
    int tx = threadIdx.x, ty = threadIdx.y;
    int t = ty * 16 + tx;
    for (int idx = t; idx < 1024; idx += 256) {
        int r = idx >> 6, c = idx & 63;
        pi[r][c] = Kuu[(size_t)(base + blockIdx.y * 16 + r) * M_TOT + b + c];
        pj[r][c] = Kuu[(size_t)(base + blockIdx.x * 16 + r) * M_TOT + b + c];
    }
    __syncthreads();
    int i = base + blockIdx.y * 16 + ty;
    int j = base + blockIdx.x * 16 + tx;
    float acc = 0.f;
#pragma unroll
    for (int k = 0; k < 64; ++k) acc += pi[ty][k] * pj[tx][k];
    Kuu[(size_t)i * M_TOT + j] -= acc;
}

// 2*sum(log|diag|) -> slot (works for Lk and for raw q_L)
__global__ void k_logdet_diag(const float* __restrict__ A, float* slot) {
    int i = blockIdx.x * 256 + threadIdx.x;
    if (i < M_TOT) atomicAdd(slot, 2.f * logf(fabsf(A[(size_t)i * M_TOT + i])));
}

// blocked triangular inverse (BS=64): block-row ib of Linv = Lk^-1
__global__ void k_trinv_step(const float* __restrict__ Lk, float* __restrict__ Linv, int ib) {
    __shared__ float Lkk[64][64];
    __shared__ float Lrow[64][65];
    int tid = threadIdx.x, b = ib * 64, C = b + 64;
    for (int idx = tid; idx < 4096; idx += 256)
        Lkk[idx >> 6][idx & 63] = Lk[(size_t)(b + (idx >> 6)) * M_TOT + b + (idx & 63)];
    __syncthreads();
    int j = blockIdx.x * 256 + tid;
    bool act = (j < C);
    float acc[64];
#pragma unroll
    for (int rr = 0; rr < 64; ++rr) acc[rr] = (act && (b + rr == j)) ? 1.f : 0.f;
    for (int c0 = 0; c0 < b; c0 += 64) {
        __syncthreads();
        for (int idx = tid; idx < 4096; idx += 256)
            Lrow[idx >> 6][idx & 63] = Lk[(size_t)(b + (idx >> 6)) * M_TOT + c0 + (idx & 63)];
        __syncthreads();
        if (act) {
            int k0 = (j > c0) ? j : c0;
            int k1 = c0 + 64;
            for (int k = k0; k < k1; ++k) {
                float lv = Linv[(size_t)k * M_TOT + j];
                int kk = k - c0;
#pragma unroll
                for (int rr = 0; rr < 64; ++rr) acc[rr] -= Lrow[rr][kk] * lv;
            }
        }
    }
    if (act) {
#pragma unroll
        for (int rr = 0; rr < 64; ++rr) {
            float v = acc[rr];
            for (int t2 = 0; t2 < rr; ++t2) v -= Lkk[rr][t2] * acc[t2];
            acc[rr] = v / Lkk[rr][rr];
            Linv[(size_t)(b + rr) * M_TOT + j] = acc[rr];
        }
    }
}

__global__ void k_tof16(const float* __restrict__ src, _Float16* __restrict__ dst, int n) {
    int i = blockIdx.x * 256 + threadIdx.x;
    if (i < n) dst[i] = (_Float16)src[i];
}

// LST[j][k] = L_S[k][j] = (k>=j) ? qL[k][j] : 0   (f16, row-major = L_S^T)
__global__ void k_lst(const float* __restrict__ qL, _Float16* __restrict__ LST) {
    int k = blockIdx.x * 16 + threadIdx.x;
    int j = blockIdx.y * 16 + threadIdx.y;
    float v = (k >= j) ? qL[(size_t)k * M_TOT + j] : 0.f;
    LST[(size_t)j * M_TOT + k] = (_Float16)v;
}

// Kfu16[n][m] = rbf(x_n, z_m) in f16
__global__ void k_kfu(const float* __restrict__ x, const float* __restrict__ z,
                      _Float16* __restrict__ Kfu) {
    __shared__ float sx[16][8], sz[16][8];
    int tx = threadIdx.x, ty = threadIdx.y;
    int m0 = blockIdx.x * 16, n0 = blockIdx.y * 16;
    if (tx < 8) { sx[ty][tx] = x[(size_t)(n0 + ty) * 8 + tx];
                  sz[ty][tx] = z[(size_t)(m0 + ty) * 8 + tx]; }
    __syncthreads();
    float d2 = 0.f;
#pragma unroll
    for (int d = 0; d < 8; ++d) { float t = sx[ty][d] - sz[tx][d]; d2 += t * t; }
    Kfu[(size_t)(n0 + ty) * M_TOT + m0 + tx] = (_Float16)(VARF * expf(NEG_HALF_INV_LS2 * d2));
}

// P = Kfu * Linv^T ; 2x2 register blocking: 32x32 output per wave.
// Linv^T upper-tri -> K loop clipped to k <= j0+31.
__global__ void k_gemm_P(const _Float16* __restrict__ A, const _Float16* __restrict__ Bt,
                         _Float16* __restrict__ P) {
    int wave = threadIdx.x >> 5, lane = threadIdx.x & 31;
    int tile = blockIdx.x * 8 + wave;
    int jt = tile & 63, rt = tile >> 6;            // (M/32) x (N/32)
    int n0 = rt * 32, j0 = jt * 32;
    v8f c00 = {}, c01 = {}, c10 = {}, c11 = {};
    for (int ks = 0; ks <= jt; ++ks) {
        int k0 = ks * 32;
        v16h a0 = load_a_frag(A, n0, k0, lane);
        v16h a1 = load_a_frag(A, n0 + 16, k0, lane);
        v16h b0 = load_bt_frag(Bt, j0, k0, lane);
        v16h b1 = load_bt_frag(Bt, j0 + 16, k0, lane);
        c00 = wmma_f16(a0, b0, c00);
        c01 = wmma_f16(a0, b1, c01);
        c10 = wmma_f16(a1, b0, c10);
        c11 = wmma_f16(a1, b1, c11);
    }
    store_tile(P, n0, j0, lane, c00);
    store_tile(P, n0, j0 + 16, lane, c01);
    store_tile(P, n0 + 16, j0, lane, c10);
    store_tile(P, n0 + 16, j0 + 16, lane, c11);
}

// W = Linv * L_S (both lower-tri); 2x2 blocking; store W^T f16; accumulate sum(W*W)
__global__ void k_gemm_W(const _Float16* __restrict__ A, const _Float16* __restrict__ Bt,
                         _Float16* __restrict__ WT, float* __restrict__ sumW2) {
    int wave = threadIdx.x >> 5, lane = threadIdx.x & 31;
    int tile = blockIdx.x * 8 + wave;
    int jt = tile & 63, it = tile >> 6;
    int i0 = it * 32, j0 = jt * 32;
    v8f c00 = {}, c01 = {}, c10 = {}, c11 = {};
    for (int ks = jt; ks <= it; ++ks) {            // j0 <= k <= i0+31
        int k0 = ks * 32;
        v16h a0 = load_a_frag(A, i0, k0, lane);
        v16h a1 = load_a_frag(A, i0 + 16, k0, lane);
        v16h b0 = load_bt_frag(Bt, j0, k0, lane);
        v16h b1 = load_bt_frag(Bt, j0 + 16, k0, lane);
        c00 = wmma_f16(a0, b0, c00);
        c01 = wmma_f16(a0, b1, c01);
        c10 = wmma_f16(a1, b0, c10);
        c11 = wmma_f16(a1, b1, c11);
    }
    float ss = 0.f;
#pragma unroll
    for (int v = 0; v < 8; ++v)
        ss += c00[v]*c00[v] + c01[v]*c01[v] + c10[v]*c10[v] + c11[v]*c11[v];
    store_tile_T(WT, i0, j0, lane, c00);
    store_tile_T(WT, i0, j0 + 16, lane, c01);
    store_tile_T(WT, i0 + 16, j0, lane, c10);
    store_tile_T(WT, i0 + 16, j0 + 16, lane, c11);
    atomicAdd(sumW2, ss);
}

// rowsum((P*W_f)^2), rowsum((P*W_g)^2): 2x2 blocking, both GPs fused on shared
// A-fragments -> 8 WMMA per 6 fragment loads.
__global__ void k_gemm_B(const _Float16* __restrict__ P, const _Float16* __restrict__ Btf,
                         const _Float16* __restrict__ Btg,
                         float* __restrict__ bbF, float* __restrict__ bbG) {
    int wave = threadIdx.x >> 5, lane = threadIdx.x & 31;
    int tile = blockIdx.x * 8 + wave;
    int jt = tile & 63, rt = tile >> 6;
    int n0 = rt * 32, j0 = jt * 32;
    v8f cf00 = {}, cf01 = {}, cf10 = {}, cf11 = {};
    v8f cg00 = {}, cg01 = {}, cg10 = {}, cg11 = {};
    for (int ks = jt; ks < 64; ++ks) {             // W lower-tri: k >= j0
        int k0 = ks * 32;
        __builtin_prefetch(P + (size_t)(n0 + (lane & 15)) * M_TOT + k0 + 64, 0, 1);
        v16h a0 = load_a_frag(P, n0, k0, lane);
        v16h a1 = load_a_frag(P, n0 + 16, k0, lane);
        v16h bf0 = load_bt_frag(Btf, j0, k0, lane);
        v16h bf1 = load_bt_frag(Btf, j0 + 16, k0, lane);
        v16h bg0 = load_bt_frag(Btg, j0, k0, lane);
        v16h bg1 = load_bt_frag(Btg, j0 + 16, k0, lane);
        cf00 = wmma_f16(a0, bf0, cf00);
        cf01 = wmma_f16(a0, bf1, cf01);
        cf10 = wmma_f16(a1, bf0, cf10);
        cf11 = wmma_f16(a1, bf1, cf11);
        cg00 = wmma_f16(a0, bg0, cg00);
        cg01 = wmma_f16(a0, bg1, cg01);
        cg10 = wmma_f16(a1, bg0, cg10);
        cg11 = wmma_f16(a1, bg1, cg11);
    }
    int kh = lane >> 4;
#pragma unroll
    for (int v = 0; v < 8; ++v) {
        float sf0 = cf00[v]*cf00[v] + cf01[v]*cf01[v];
        float sf1 = cf10[v]*cf10[v] + cf11[v]*cf11[v];
        float sg0 = cg00[v]*cg00[v] + cg01[v]*cg01[v];
        float sg1 = cg10[v]*cg10[v] + cg11[v]*cg11[v];
        for (int m = 1; m < 16; m <<= 1) {
            sf0 += __shfl_xor(sf0, m, 32);
            sf1 += __shfl_xor(sf1, m, 32);
            sg0 += __shfl_xor(sg0, m, 32);
            sg1 += __shfl_xor(sg1, m, 32);
        }
        if ((lane & 15) == 0) {
            int r = n0 + v + 8 * kh;
            atomicAdd(bbF + r, sf0);
            atomicAdd(bbF + r + 16, sf1);
            atomicAdd(bbG + r, sg0);
            atomicAdd(bbG + r + 16, sg1);
        }
    }
}

__global__ void k_sumpp(const _Float16* __restrict__ P, float* __restrict__ out) {
    int n = blockIdx.x * 256 + threadIdx.x;
    const v8h* row = (const v8h*)(P + (size_t)n * M_TOT);
    float s = 0.f;
    for (int j = 0; j < M_TOT / 8; ++j) {
        v8h p = row[j];
#pragma unroll
        for (int i = 0; i < 8; ++i) { float pv = (float)p[i]; s += pv * pv; }
    }
    out[n] = s;
}

__global__ void k_alpha(const float* __restrict__ Linv, const float* __restrict__ qm,
                        float* __restrict__ alpha, float* __restrict__ sumA2) {
    int i = blockIdx.x * 256 + threadIdx.x;
    if (i >= M_TOT) return;
    float s = 0.f;
    for (int k = 0; k <= i; ++k) s += Linv[(size_t)i * M_TOT + k] * qm[k];
    alpha[i] = s;
    atomicAdd(sumA2, s * s);
}

__global__ void k_gemv_m(const _Float16* __restrict__ P, const float* __restrict__ af,
                         const float* __restrict__ ag,
                         float* __restrict__ mf, float* __restrict__ mg) {
    __shared__ float saf[M_TOT];
    __shared__ float sag[M_TOT];
    for (int idx = threadIdx.x; idx < M_TOT; idx += 256) { saf[idx] = af[idx]; sag[idx] = ag[idx]; }
    __syncthreads();
    int n = blockIdx.x * 256 + threadIdx.x;
    const v8h* row = (const v8h*)(P + (size_t)n * M_TOT);
    float f = 0.f, g = 0.f;
    for (int j = 0; j < M_TOT / 8; ++j) {
        v8h p = row[j];
#pragma unroll
        for (int i = 0; i < 8; ++i) {
            float pv = (float)p[i];
            f += pv * saf[j * 8 + i];
            g += pv * sag[j * 8 + i];
        }
    }
    mf[n] = f; mg[n] = g;
}

__global__ void k_expect(const float* __restrict__ y, const float* __restrict__ mf,
                         const float* __restrict__ mg, const float* __restrict__ pp,
                         const float* __restrict__ bbf, const float* __restrict__ bbg,
                         float* slot) {
    int n = blockIdx.x * 256 + threadIdx.x;
    float vf = VARF + bbf[n] - pp[n];
    float vg = VARF + bbg[n] - pp[n];
    float dy = y[n] - mf[n];
    float e = -HALF_LOG_2PI_F - 0.5f * mg[n]
              - 0.5f * (dy * dy + vf) * expf(-mg[n] + 0.5f * vg);
    atomicAdd(slot, e);
}

// slots: 0 logdetK, 1 logdetS_f, 2 logdetS_g, 3 sumW2_f, 4 sumW2_g,
//        5 sumA2_f, 6 sumA2_g, 7 expsum
__global__ void k_finalize(const float* __restrict__ s, float* __restrict__ out) {
    float klf = 0.5f * (s[3] + s[5] - (float)M_TOT + s[0] - s[1]);
    float klg = 0.5f * (s[4] + s[6] - (float)M_TOT + s[0] - s[2]);
    out[0] = -(s[7] - (klf + klg));
}

extern "C" void kernel_launch(void* const* d_in, const int* in_sizes, int n_in,
                              void* d_out, int out_size, void* d_ws, size_t ws_size,
                              hipStream_t stream) {
    (void)in_sizes; (void)n_in; (void)out_size; (void)ws_size;
    const float* x   = (const float*)d_in[0];
    const float* y   = (const float*)d_in[1];
    const float* z   = (const float*)d_in[2];
    const float* qmf = (const float*)d_in[3];
    const float* qLf = (const float*)d_in[4];
    const float* qmg = (const float*)d_in[5];
    const float* qLg = (const float*)d_in[6];
    float* out = (float*)d_out;

    const size_t MM = (size_t)M_TOT * M_TOT;
    const size_t NM = (size_t)N_TOT * M_TOT;

    char* w = (char*)d_ws;
    float*    Kuu    = (float*)w;      w += MM * 4;     // becomes Lk in place
    float*    Linv   = (float*)w;      w += MM * 4;
    _Float16* L16    = (_Float16*)w;   w += MM * 2;     // Linv in f16 (row-major)
    _Float16* LST16  = (_Float16*)w;   w += MM * 2;     // L_S^T f16 (reused f then g)
    _Float16* WT16f  = (_Float16*)w;   w += MM * 2;
    _Float16* WT16g  = (_Float16*)w;   w += MM * 2;
    _Float16* Kfu16  = (_Float16*)w;   w += NM * 2;
    _Float16* P16    = (_Float16*)w;   w += NM * 2;
    float*    sumPP  = (float*)w;      w += (size_t)N_TOT * 4;
    float*    sumBBf = (float*)w;      w += (size_t)N_TOT * 4;
    float*    sumBBg = (float*)w;      w += (size_t)N_TOT * 4;
    float*    m_f    = (float*)w;      w += (size_t)N_TOT * 4;
    float*    m_g    = (float*)w;      w += (size_t)N_TOT * 4;
    float*    alphaf = (float*)w;      w += (size_t)M_TOT * 4;
    float*    alphag = (float*)w;      w += (size_t)M_TOT * 4;
    float*    slots  = (float*)w;      w += 16 * 4;

    // ---- zero init ----
    k_fill0<<<(int)((MM + 255) / 256), 256, 0, stream>>>(Linv, (int)MM);
    k_fill0<<<(N_TOT + 255) / 256, 256, 0, stream>>>(sumBBf, N_TOT);
    k_fill0<<<(N_TOT + 255) / 256, 256, 0, stream>>>(sumBBg, N_TOT);
    k_fill0<<<1, 256, 0, stream>>>(slots, 16);

    // ---- Kuu + blocked Cholesky (BS=64, in place) ----
    k_kuu<<<dim3(M_TOT / 16, M_TOT / 16), dim3(16, 16), 0, stream>>>(z, Kuu);
    for (int kb = 0; kb < M_TOT / 64; ++kb) {
        k_chol_diag<<<1, 256, 0, stream>>>(Kuu, kb);
        int R = M_TOT - (kb + 1) * 64;
        if (R > 0) {
            k_chol_panel<<<(R + 255) / 256, 256, 0, stream>>>(Kuu, kb);
            k_chol_syrk<<<dim3(R / 16, R / 16), dim3(16, 16), 0, stream>>>(Kuu, kb);
        }
    }
    k_logdet_diag<<<M_TOT / 256, 256, 0, stream>>>(Kuu, slots + 0);

    // ---- triangular inverse Linv = Lk^-1 (BS=64) ----
    for (int ib = 0; ib < M_TOT / 64; ++ib) {
        int C = (ib + 1) * 64;
        k_trinv_step<<<(C + 255) / 256, 256, 0, stream>>>(Kuu, Linv, ib);
    }
    k_tof16<<<(int)((MM + 255) / 256), 256, 0, stream>>>(Linv, L16, (int)MM);

    // ---- Kfu and big GEMM P = Kfu * Linv^T (WMMA f16->f32) ----
    k_kfu<<<dim3(M_TOT / 16, N_TOT / 16), dim3(16, 16), 0, stream>>>(x, z, Kfu16);
    k_gemm_P<<<(N_TOT / 32) * (M_TOT / 32) / 8, 256, 0, stream>>>(Kfu16, L16, P16);
    k_sumpp<<<N_TOT / 256, 256, 0, stream>>>(P16, sumPP);

    // ---- per-GP small pieces ----
    k_lst<<<dim3(M_TOT / 16, M_TOT / 16), dim3(16, 16), 0, stream>>>(qLf, LST16);
    k_logdet_diag<<<M_TOT / 256, 256, 0, stream>>>(qLf, slots + 1);
    k_gemm_W<<<(M_TOT / 32) * (M_TOT / 32) / 8, 256, 0, stream>>>(L16, LST16, WT16f, slots + 3);
    k_alpha<<<M_TOT / 256, 256, 0, stream>>>(Linv, qmf, alphaf, slots + 5);

    k_lst<<<dim3(M_TOT / 16, M_TOT / 16), dim3(16, 16), 0, stream>>>(qLg, LST16);
    k_logdet_diag<<<M_TOT / 256, 256, 0, stream>>>(qLg, slots + 2);
    k_gemm_W<<<(M_TOT / 32) * (M_TOT / 32) / 8, 256, 0, stream>>>(L16, LST16, WT16g, slots + 4);
    k_alpha<<<M_TOT / 256, 256, 0, stream>>>(Linv, qmg, alphag, slots + 6);

    // ---- fused big GEMM: rowsum((P W_f)^2), rowsum((P W_g)^2) ----
    k_gemm_B<<<(N_TOT / 32) * (M_TOT / 32) / 8, 256, 0, stream>>>(P16, WT16f, WT16g, sumBBf, sumBBg);

    // ---- means, expectation, final scalar ----
    k_gemv_m<<<N_TOT / 256, 256, 0, stream>>>(P16, alphaf, alphag, m_f, m_g);
    k_expect<<<N_TOT / 256, 256, 0, stream>>>(y, m_f, m_g, sumPP, sumBBf, sumBBg, slots + 7);
    k_finalize<<<1, 1, 0, stream>>>(slots, out);
}
